// DeepCrossModalAlign_2602750181431
// MI455X (gfx1250) — matrix operand order repaired and checked by reading, not compiled, not run on hardware
//
#include <hip/hip_runtime.h>
#include <hip/hip_bf16.h>

// ---------------------------------------------------------------------------
// Problem dims (DeepCrossModalAlign)
// ---------------------------------------------------------------------------
#define B_  16
#define LP_ 512
#define LA_ 32
#define DP_ 1024
#define DA_ 512
#define H_  512
#define DO_ 1024
#define DC_ (DP_ + DA_)   // 1536 concat dim

typedef __attribute__((ext_vector_type(16))) __bf16 v16bf;
typedef __attribute__((ext_vector_type(8)))  float  v8f;
typedef __attribute__((ext_vector_type(4)))  unsigned int u32x4;
typedef __attribute__((ext_vector_type(4)))  float  f32x4;

union FragBF16 { v16bf v; u32x4 q[2]; unsigned short u[16]; };

// round-to-nearest-even f32 -> bf16 (as raw u16)
__device__ __forceinline__ unsigned short f2bf(float f) {
    unsigned int u = __float_as_uint(f);
    u += 0x7FFFu + ((u >> 16) & 1u);
    return (unsigned short)(u >> 16);
}

// CDNA5 has V_TANH_F32 (TRANS unit). Use the native builtin if this clang
// declares it; otherwise fall back to the exp formulation (v_exp_f32 path).
#if defined(__has_builtin)
#if __has_builtin(__builtin_amdgcn_tanhf)
#define HAVE_NATIVE_TANH 1
#endif
#endif

__device__ __forceinline__ float fast_tanh(float x) {
#ifdef HAVE_NATIVE_TANH
    return __builtin_amdgcn_tanhf(x);
#else
    // tanh(x) = 1 - 2/(exp(2x)+1) ; monotone-correct at +-inf
    return 1.0f - 2.0f / (1.0f + __expf(2.0f * x));
#endif
}

// ---------------------------------------------------------------------------
// Elementwise f32 -> bf16 (n multiple of 4)
// ---------------------------------------------------------------------------
__global__ __launch_bounds__(256) void cvt_bf16_kernel(const float* __restrict__ in,
                                                       unsigned short* __restrict__ out,
                                                       long long n4) {
    long long i = ((long long)blockIdx.x * blockDim.x + threadIdx.x);
    if (i >= n4) return;
    f32x4 v = *(const f32x4*)(in + 4 * i);
    unsigned short r0 = f2bf(v[0]), r1 = f2bf(v[1]), r2 = f2bf(v[2]), r3 = f2bf(v[3]);
    unsigned int lo = (unsigned int)r0 | ((unsigned int)r1 << 16);
    unsigned int hi = (unsigned int)r2 | ((unsigned int)r3 << 16);
    *(uint2*)(out + 4 * i) = make_uint2(lo, hi);
}

// ---------------------------------------------------------------------------
// Transpose + convert: out[n*Kd + k] = (bf16) in[k*Nd + n]   (batched via z)
// ---------------------------------------------------------------------------
__global__ __launch_bounds__(256) void transpose_bf16_kernel(const float* __restrict__ in,
                                                             unsigned short* __restrict__ out,
                                                             int Kd, int Nd,
                                                             long long sIn, long long sOut) {
    in  += (long long)blockIdx.z * sIn;
    out += (long long)blockIdx.z * sOut;
    int idx = blockIdx.x * blockDim.x + threadIdx.x;
    if (idx >= Kd * Nd) return;
    int k = idx / Nd;
    int n = idx - k * Nd;
    out[(size_t)n * Kd + k] = f2bf(in[(size_t)k * Nd + n]);
}

// ---------------------------------------------------------------------------
// WMMA fragment loaders (wave32, v_wmma_f32_16x16x32_bf16 layouts)
//   A  16x32 (MxK): lane m=L&15, ksel=L>>4; elems 0..7 -> K = 8*ksel+0..7,
//                   elems 8..15 -> K = 16+8*ksel+0..7   (two b128 loads)
//   B  32x16 (KxN) given as BT[N,K]: lane n=L&15, ksel=L>>4;
//                   elems 0..7 -> K = 8*ksel+0..7,
//                   elems 8..15 -> K = 16+8*ksel+0..7   (two b128 loads)
// ---------------------------------------------------------------------------
__device__ __forceinline__ v16bf load_a_frag(const unsigned short* __restrict__ A,
                                             int lda, int row0, int kk, int lane) {
    int m = lane & 15, ks = lane >> 4;
    const unsigned short* p = A + (size_t)(row0 + m) * lda + kk + 8 * ks;
    FragBF16 f;
    f.q[0] = *(const u32x4*)(p);
    f.q[1] = *(const u32x4*)(p + 16);
    return f.v;
}

__device__ __forceinline__ v16bf load_b_frag(const unsigned short* __restrict__ BT,
                                             int ldbt, int col0, int kk, int lane) {
    int n = lane & 15, ks = lane >> 4;
    const unsigned short* p = BT + (size_t)(col0 + n) * ldbt + kk + 8 * ks;
    FragBF16 f;
    f.q[0] = *(const u32x4*)(p);
    f.q[1] = *(const u32x4*)(p + 16);
    return f.v;
}

// ---------------------------------------------------------------------------
// Generic WMMA GEMM:  C[M,N] = A[M,K] (bf16) * BT[N,K]^T (bf16)  (+bias)(relu)
//   block = 256 threads = 8 waves, waves tiled 2(M) x 4(N), wave tile 32x64
//   block tile = 64(M) x 256(N). M%64==0, N%256==0, K%32==0 required.
//   CONCAT: A covers k<K1 (lda), A2 covers k>=K1 (lda2); K1%32==0.
//   DUAL:   also emit bf16 copy of C into Cbf.
//   Batched via blockIdx.z with element strides sA/sA2/sBT/sC.
// ---------------------------------------------------------------------------
template <bool BIAS, bool RELU, bool CONCAT, bool DUAL>
__global__ __launch_bounds__(256) void gemm_wmma_kernel(
        const unsigned short* __restrict__ A,
        const unsigned short* __restrict__ A2,
        const unsigned short* __restrict__ BT,
        const float* __restrict__ bias,
        float* __restrict__ C,
        unsigned short* __restrict__ Cbf,
        int K, int K1,
        int lda, int lda2, int ldbt, int ldc,
        long long sA, long long sA2, long long sBT, long long sC) {
    const int lane = threadIdx.x & 31;
    const int wave = threadIdx.x >> 5;
    const int wm = wave & 1;        // 0..1
    const int wn = wave >> 1;       // 0..3
    const long long z = blockIdx.z;

    A  += z * sA;
    BT += z * sBT;
    C  += z * sC;
    if (CONCAT) A2 += z * sA2;
    if (DUAL)   Cbf += z * sC;

    const int row0 = blockIdx.x * 64 + wm * 32;
    const int col0 = blockIdx.y * 256 + wn * 64;

    v8f acc[2][4];
#pragma unroll
    for (int mt = 0; mt < 2; ++mt)
#pragma unroll
        for (int nt = 0; nt < 4; ++nt) acc[mt][nt] = (v8f)(0.0f);

    for (int kk = 0; kk < K; kk += 32) {
        const unsigned short* Asrc = A;
        int ldA = lda, kcol = kk;
        if (CONCAT && kk >= K1) { Asrc = A2; ldA = lda2; kcol = kk - K1; }

        v16bf a0 = load_a_frag(Asrc, ldA, row0,      kcol, lane);
        v16bf a1 = load_a_frag(Asrc, ldA, row0 + 16, kcol, lane);
#pragma unroll
        for (int nt = 0; nt < 4; ++nt) {
            v16bf b = load_b_frag(BT, ldbt, col0 + nt * 16, kk, lane);
            acc[0][nt] = __builtin_amdgcn_wmma_f32_16x16x32_bf16(
                false, a0, false, b, (short)0, acc[0][nt], false, false);
            acc[1][nt] = __builtin_amdgcn_wmma_f32_16x16x32_bf16(
                false, a1, false, b, (short)0, acc[1][nt], false, false);
        }
    }

    // Epilogue. C/D layout: lane n=L&15, msel=L>>4; VGPR i -> row = 8*msel + i.
    const int n  = lane & 15;
    const int ms = lane >> 4;
#pragma unroll
    for (int mt = 0; mt < 2; ++mt) {
#pragma unroll
        for (int nt = 0; nt < 4; ++nt) {
            const int col = col0 + nt * 16 + n;
            float bv = 0.0f;
            if (BIAS) bv = bias[col];
#pragma unroll
            for (int i = 0; i < 8; ++i) {
                int r = row0 + mt * 16 + 8 * ms + i;
                float v = acc[mt][nt][i] + bv;
                if (RELU) v = fmaxf(v, 0.0f);
                C[(size_t)r * ldc + col] = v;
                if (DUAL) Cbf[(size_t)r * ldc + col] = f2bf(v);
            }
        }
    }
}

// ---------------------------------------------------------------------------
// Fused Bahdanau scores + mask + softmax over LA=32 (one lane per answer slot)
//   scores[b,p,a] = sum_h ww[h]*tanh(pplAlign[b,p,h] + ansAlign[b,a,h]) + wb
//   block = 256 = 8 waves; each wave handles one (b,p) row; lane = a.
// ---------------------------------------------------------------------------
__global__ __launch_bounds__(256) void scores_softmax_kernel(
        const float* __restrict__ pplAlign,   // [B*LP, H]
        const float* __restrict__ ansAlign,   // [B*LA, H]
        const int*   __restrict__ amask,      // [B, LA]
        const float* __restrict__ ww,         // [H]
        const float* __restrict__ wb,         // [1]
        unsigned short* __restrict__ attnB) { // [B*LP, LA] bf16
    __shared__ float sWW[H_];
    __shared__ float sPA[8][H_];

    const int lane = threadIdx.x & 31;
    const int wave = threadIdx.x >> 5;

    for (int i = threadIdx.x; i < H_; i += 256) sWW[i] = ww[i];

    const int row = blockIdx.x * 8 + wave;      // b*LP + p
    const int b   = row >> 9;                   // / LP_
    const float* paRow = pplAlign + (size_t)row * H_;
    float* sp = sPA[wave];
    for (int i = lane; i < H_; i += 32) sp[i] = paRow[i];
    __syncthreads();

    const int a = lane;
    const float* ansRow = ansAlign + (size_t)(b * LA_ + a) * H_;

    float acc = 0.0f;
#pragma unroll 4
    for (int h = 0; h < H_; h += 4) {
        f32x4 av = *(const f32x4*)(ansRow + h);
        f32x4 pv = *(const f32x4*)(sp + h);
        f32x4 wv = *(const f32x4*)(sWW + h);
        acc += wv[0] * fast_tanh(pv[0] + av[0]);
        acc += wv[1] * fast_tanh(pv[1] + av[1]);
        acc += wv[2] * fast_tanh(pv[2] + av[2]);
        acc += wv[3] * fast_tanh(pv[3] + av[3]);
    }
    float score = acc + wb[0];
    if (amask[b * LA_ + a] == 0) score = -1e8f;

    // softmax across the 32 lanes (LA == wave32)
    float mx = score;
#pragma unroll
    for (int off = 16; off >= 1; off >>= 1) mx = fmaxf(mx, __shfl_xor(mx, off, 32));
    float e = __expf(score - mx);
    float sm = e;
#pragma unroll
    for (int off = 16; off >= 1; off >>= 1) sm += __shfl_xor(sm, off, 32);
    float attn = e / sm;

    attnB[(size_t)row * LA_ + a] = f2bf(attn);
}

// ---------------------------------------------------------------------------
// Launch
// ---------------------------------------------------------------------------
extern "C" void kernel_launch(void* const* d_in, const int* in_sizes, int n_in,
                              void* d_out, int out_size, void* d_ws, size_t ws_size,
                              hipStream_t stream) {
    (void)in_sizes; (void)n_in; (void)out_size; (void)ws_size;

    const float* ppl    = (const float*)d_in[0];
    /* loc (d_in[1]) unused by the reference forward */
    const float* answer = (const float*)d_in[2];
    const int*   amask  = (const int*)  d_in[3];
    const float* Wp     = (const float*)d_in[4];
    const float* bp     = (const float*)d_in[5];
    const float* Wa     = (const float*)d_in[6];
    const float* ba     = (const float*)d_in[7];
    const float* ww     = (const float*)d_in[8];
    const float* wb     = (const float*)d_in[9];
    const float* Wo     = (const float*)d_in[10];
    const float* bo     = (const float*)d_in[11];

    float* out = (float*)d_out;
    const long long ALIGNED_ELEMS = (long long)B_ * LP_ * DO_;   // 8,388,608
    float* outAgg = out + ALIGNED_ELEMS;                         // [B,LP,DA]

    // ---- workspace carve ----
    char* w = (char*)d_ws;
    auto carve = [&](size_t bytes) { char* p = w; w += (bytes + 255) & ~(size_t)255; return p; };
    unsigned short* WpT   = (unsigned short*)carve((size_t)H_ * DP_ * 2);      // [H, DP]
    unsigned short* WaT   = (unsigned short*)carve((size_t)H_ * DA_ * 2);      // [H, DA]
    unsigned short* WoT   = (unsigned short*)carve((size_t)DO_ * DC_ * 2);     // [DO, DC]
    unsigned short* pplB  = (unsigned short*)carve((size_t)B_ * LP_ * DP_ * 2);// [B*LP, DP]
    unsigned short* ansB  = (unsigned short*)carve((size_t)B_ * LA_ * DA_ * 2);// [B*LA, DA]
    unsigned short* ansT  = (unsigned short*)carve((size_t)B_ * DA_ * LA_ * 2);// [B][DA, LA]
    unsigned short* attnB = (unsigned short*)carve((size_t)B_ * LP_ * LA_ * 2);// [B*LP, LA]
    unsigned short* aggB  = (unsigned short*)carve((size_t)B_ * LP_ * DA_ * 2);// [B*LP, DA]
    float* pplAlign = (float*)carve((size_t)B_ * LP_ * H_ * 4);                // [B*LP, H]
    float* ansAlign = (float*)carve((size_t)B_ * LA_ * H_ * 4);                // [B*LA, H]

    const dim3 blk(256);

    // ---- weight transposes (f32 -> bf16 BT layout) ----
    transpose_bf16_kernel<<<dim3((DP_ * H_ + 255) / 256, 1, 1), blk, 0, stream>>>(
        Wp, WpT, DP_, H_, 0, 0);
    transpose_bf16_kernel<<<dim3((DA_ * H_ + 255) / 256, 1, 1), blk, 0, stream>>>(
        Wa, WaT, DA_, H_, 0, 0);
    transpose_bf16_kernel<<<dim3((DC_ * DO_ + 255) / 256, 1, 1), blk, 0, stream>>>(
        Wo, WoT, DC_, DO_, 0, 0);
    // per-batch answer transpose: ansT[b][d, a] = answer[b][a, d]
    transpose_bf16_kernel<<<dim3((LA_ * DA_ + 255) / 256, 1, B_), blk, 0, stream>>>(
        answer, ansT, LA_, DA_, (long long)LA_ * DA_, (long long)LA_ * DA_);

    // ---- activation conversions ----
    cvt_bf16_kernel<<<dim3((unsigned)(((long long)B_ * LP_ * DP_ / 4 + 255) / 256)), blk, 0, stream>>>(
        ppl, pplB, (long long)B_ * LP_ * DP_ / 4);
    cvt_bf16_kernel<<<dim3((unsigned)(((long long)B_ * LA_ * DA_ / 4 + 255) / 256)), blk, 0, stream>>>(
        answer, ansB, (long long)B_ * LA_ * DA_ / 4);

    // ---- GEMM1: pplAlign[8192,512] = pplB[8192,1024] @ WpT^T + bp ----
    gemm_wmma_kernel<true, false, false, false>
        <<<dim3(B_ * LP_ / 64, H_ / 256, 1), blk, 0, stream>>>(
            pplB, nullptr, WpT, bp, pplAlign, nullptr,
            DP_, 0, DP_, 0, DP_, H_, 0, 0, 0, 0);

    // ---- GEMM2: ansAlign[512,512] = ansB[512,512] @ WaT^T + ba ----
    gemm_wmma_kernel<true, false, false, false>
        <<<dim3(B_ * LA_ / 64, H_ / 256, 1), blk, 0, stream>>>(
            ansB, nullptr, WaT, ba, ansAlign, nullptr,
            DA_, 0, DA_, 0, DA_, H_, 0, 0, 0, 0);

    // ---- fused tanh-scores + mask + softmax -> attnB (bf16) ----
    scores_softmax_kernel<<<dim3(B_ * LP_ / 8), blk, 0, stream>>>(
        pplAlign, ansAlign, amask, ww, wb, attnB);

    // ---- agg GEMM (batched, K=32): agg[b][512,512] = attn[b] @ answer[b] ----
    //      dual output: f32 -> d_out tail, bf16 -> aggB for final GEMM
    gemm_wmma_kernel<false, false, false, true>
        <<<dim3(LP_ / 64, DA_ / 256, B_), blk, 0, stream>>>(
            attnB, nullptr, ansT, nullptr, outAgg, aggB,
            LA_, 0, LA_, 0, LA_, DA_,
            (long long)LP_ * LA_, 0, (long long)DA_ * LA_, (long long)LP_ * DA_);

    // ---- final concat GEMM: aligned = relu([pplB | aggB] @ WoT^T + bo) ----
    gemm_wmma_kernel<true, true, true, false>
        <<<dim3(B_ * LP_ / 64, DO_ / 256, 1), blk, 0, stream>>>(
            pplB, aggB, WoT, bo, out, nullptr,
            DC_, DP_, DP_, DA_, DC_, DO_, 0, 0, 0, 0);
}
